// LSTM_80470507258143
// MI455X (gfx1250) — matrix operand order repaired and checked by reading, not compiled, not run on hardware
//
#include <hip/hip_runtime.h>

typedef __attribute__((ext_vector_type(8)))  _Float16 v8h;
typedef __attribute__((ext_vector_type(16))) _Float16 v16h;
typedef __attribute__((ext_vector_type(8)))  float    v8f;

constexpr int kH     = 512;
constexpr int kG     = 4 * kH;        // 2048 gate rows
constexpr int kB     = 32;
constexpr int kT     = 1024;
constexpr int kBH    = kB * kH;       // 16384 elements per time slot
constexpr int kNWG   = 16;            // persistent workgroups
constexpr int kUPW   = 32;            // hidden units per workgroup
constexpr int kNCOL  = 4 * kUPW;      // 128 gate columns per WG

struct HPair { v8h lo, hi; };
static __device__ __forceinline__ v16h mk16(v8h lo, v8h hi) {
  HPair p{lo, hi};
  return __builtin_bit_cast(v16h, p);
}
static __device__ __forceinline__ float sigf(float x) {
  return 1.0f / (1.0f + __expf(-x));
}

// ---------------------------------------------------------------------------
// Prep: fp32->fp16 weight conversion (layer1 concatenates W_ih|W_hh along K),
// bias fusion, initial hidden state, barrier counter reset.
// ---------------------------------------------------------------------------
__global__ void prep_kernel(
    const float* __restrict__ Whh0, const float* __restrict__ Wih1,
    const float* __restrict__ Whh1, const float* __restrict__ Wih0,
    const float* __restrict__ bih0, const float* __restrict__ bhh0,
    const float* __restrict__ bih1, const float* __restrict__ bhh1,
    const float* __restrict__ h0,
    _Float16* __restrict__ wbuf0, _Float16* __restrict__ wbuf1,
    float* __restrict__ bsum0, float* __restrict__ bsum1,
    float* __restrict__ wih0c,
    _Float16* __restrict__ h0hist, _Float16* __restrict__ h1hist,
    unsigned* __restrict__ counters)
{
  const int stride = gridDim.x * blockDim.x;
  const int tid = blockIdx.x * blockDim.x + threadIdx.x;

  for (int i = tid; i < kG * kH; i += stride)
    wbuf0[i] = (_Float16)Whh0[i];

  for (int i = tid; i < kG * 2 * kH; i += stride) {
    int n = i >> 10, k = i & 1023;
    float v = (k < kH) ? Wih1[n * kH + k] : Whh1[n * kH + (k - kH)];
    wbuf1[i] = (_Float16)v;
  }
  for (int i = tid; i < kG; i += stride) {
    bsum0[i] = bih0[i] + bhh0[i];
    bsum1[i] = bih1[i] + bhh1[i];
    wih0c[i] = Wih0[i];
  }
  for (int i = tid; i < kBH; i += stride) {
    h0hist[i] = (_Float16)h0[i];
    h1hist[i] = (_Float16)h0[kBH + i];
  }
  if (tid == 0) { counters[0] = 0u; counters[64] = 0u; }
}

// ---------------------------------------------------------------------------
// Persistent recurrent scan. 16 WGs, each owns 32 hidden units (128 gate
// columns). Weight slice lives in LDS for the whole sequence. Per step:
// WMMA GEMM (A = h history fp16 from L2, B = LDS weights), gate nonlinearity,
// fp16 h write, global-atomic grid barrier.
// kItersX = 16 for layer 1 (fused input projection, K=1024), 0 for layer 0.
// ---------------------------------------------------------------------------
__global__ void __launch_bounds__(256)
lstm_scan_kernel(const _Float16* __restrict__ wbuf,   // [2048][Klen] fp16
                 const float* __restrict__ bsum,      // [2048]
                 const float* __restrict__ wih0c,     // [2048] (layer0) or null
                 const float* __restrict__ x,         // [B][T] (layer0) or null
                 const _Float16* __restrict__ xhist,  // [T+1][B][H] (layer1)
                 _Float16* __restrict__ hist,         // [T+1][B][H]
                 const float* __restrict__ c0,        // [B][H] for this layer
                 unsigned* __restrict__ cnt,
                 int Klen, int kItersX)
{
  extern __shared__ __align__(32) char smem[];
  float*    gates = (float*)smem;                 // [128 cols][32 b] fp32
  float*    cL    = gates + kNCOL * kB;           // [32 u][32 b] cell state
  _Float16* wL    = (_Float16*)(cL + kUPW * kB);  // [128 rows][Klen] fp16

  const int wg    = blockIdx.x;
  const int unit0 = wg * kUPW;
  const int tid   = threadIdx.x;
  const int wave  = tid >> 5;
  const int lane  = tid & 31;
  const int hi    = lane >> 4;
  const int l16   = lane & 15;

  // One-time: stage this WG's weight slice into LDS (row-major, K contiguous —
  // exactly the B-matrix lane striping WMMA wants).
  {
    const int chunksPerRow = Klen / 8;
    for (int i = tid; i < kNCOL * chunksPerRow; i += blockDim.x) {
      int lrow = i / chunksPerRow;
      int ch   = i - lrow * chunksPerRow;
      int grow = (lrow >> 5) * kH + unit0 + (lrow & 31);  // gate*512 + unit
      *(v8h*)(wL + (size_t)lrow * Klen + ch * 8) =
          *(const v8h*)(wbuf + (size_t)grow * Klen + ch * 8);
    }
  }
  for (int i = tid; i < kUPW * kB; i += blockDim.x) {
    int u = i >> 5, b = i & 31;
    cL[i] = c0[b * kH + unit0 + u];
  }
  __syncthreads();

  const int kIters = Klen / 32;
  const int lrowB  = wave * 16 + l16;   // weight row (N col) this lane serves

  for (int t = 0; t < kT; ++t) {
    const _Float16* hcur = hist + (size_t)t * kBH;
    const _Float16* xcur = kItersX ? (xhist + (size_t)(t + 1) * kBH) : hcur;

    if (kItersX && (t + 2) <= kT)   // prefetch next step's input rows into L2/L0
      __builtin_prefetch(xhist + (size_t)(t + 2) * kBH + (tid << 6), 0, 0);

    v8f acc0 = {};
    v8f acc1 = {};
    for (int kk = 0; kk < kIters; ++kk) {
      const int k0 = kk * 32;
      const _Float16* Ap;
      int ka;
      if (kk < kItersX) { Ap = xcur; ka = k0; }
      else              { Ap = hcur; ka = k0 - kItersX * 32; }

      // A fragments (16-bit A 16x32 striping: lane<16 holds K {0..7,16..23},
      // lane>=16 holds K {8..15,24..31} of its row).
      v8h a0lo = *(const v8h*)(Ap + (size_t)l16 * kH + ka + hi * 8);
      v8h a0hi = *(const v8h*)(Ap + (size_t)l16 * kH + ka + 16 + hi * 8);
      v8h a1lo = *(const v8h*)(Ap + (size_t)(16 + l16) * kH + ka + hi * 8);
      v8h a1hi = *(const v8h*)(Ap + (size_t)(16 + l16) * kH + ka + 16 + hi * 8);
      // B fragment: lane n holds column n, K {0..7,16..23} (lane<16) or {8..15,24..31}.
      v8h blo  = *(const v8h*)(wL + (size_t)lrowB * Klen + k0 + hi * 8);
      v8h bhi  = *(const v8h*)(wL + (size_t)lrowB * Klen + k0 + 16 + hi * 8);

      v16h A0 = mk16(a0lo, a0hi);
      v16h A1 = mk16(a1lo, a1hi);
      v16h Bf = mk16(blo, bhi);

      acc0 = __builtin_amdgcn_wmma_f32_16x16x32_f16(false, A0, false, Bf,
                                                    (short)0, acc0, false, false);
      acc1 = __builtin_amdgcn_wmma_f32_16x16x32_f16(false, A1, false, Bf,
                                                    (short)0, acc1, false, false);
    }

    // Spill accumulators to LDS: D lane n0 holds col n0, VGPR r -> M = r+8*hi
    // => each lane's v8f is 8 contiguous batches in a [col][b] layout.
    const int gcol = wave * 16 + l16;
    *(v8f*)(gates + gcol * kB + hi * 8)      = acc0;
    *(v8f*)(gates + gcol * kB + 16 + hi * 8) = acc1;
    __syncthreads();

    // Elementwise LSTM cell: 4 (unit,batch) pairs per thread.
    {
      const int u  = tid >> 3;
      const int b0 = (tid & 7) << 2;
      const int ug = unit0 + u;
      for (int j = 0; j < 4; ++j) {
        const int b = b0 + j;
        float gi = gates[(u)            * kB + b] + bsum[ug];
        float gf = gates[(kUPW + u)     * kB + b] + bsum[kH + ug];
        float gg = gates[(2 * kUPW + u) * kB + b] + bsum[2 * kH + ug];
        float go = gates[(3 * kUPW + u) * kB + b] + bsum[3 * kH + ug];
        if (!kItersX) {  // layer 0: I==1, fold x @ W_ih in as a scalar FMA
          float xv = x[b * kT + t];
          gi += xv * wih0c[ug];
          gf += xv * wih0c[kH + ug];
          gg += xv * wih0c[2 * kH + ug];
          go += xv * wih0c[3 * kH + ug];
        }
        float c = cL[u * kB + b];
        c = sigf(gf) * c + sigf(gi) * tanhf(gg);
        float h = sigf(go) * tanhf(c);
        cL[u * kB + b] = c;
        hist[(size_t)(t + 1) * kBH + b * kH + ug] = (_Float16)h;
      }
    }

    // Grid-wide step barrier (monotonic counter; h[t+1] slices now visible).
    __threadfence();
    __syncthreads();
    if (tid == 0) {
      atomicAdd(cnt, 1u);
      const unsigned tgt = (unsigned)((t + 1) * kNWG);
      while (atomicAdd(cnt, 0u) < tgt) __builtin_amdgcn_s_sleep(1);
    }
    __syncthreads();
  }
}

// ---------------------------------------------------------------------------
// Final FC: out[b,t,:3] = h1[t] @ fc_w.T + fc_b
// ---------------------------------------------------------------------------
__global__ void fc_kernel(const _Float16* __restrict__ h1hist,
                          const float* __restrict__ fcw,
                          const float* __restrict__ fcb,
                          float* __restrict__ out)
{
  const int idx = blockIdx.x * blockDim.x + threadIdx.x;
  if (idx >= kB * kT) return;
  const int b = idx >> 10;
  const int t = idx & (kT - 1);
  const _Float16* hr = h1hist + (size_t)(t + 1) * kBH + b * kH;
  float s0 = fcb[0], s1 = fcb[1], s2 = fcb[2];
  for (int k = 0; k < kH; ++k) {
    float hv = (float)hr[k];
    s0 += hv * fcw[k];
    s1 += hv * fcw[kH + k];
    s2 += hv * fcw[2 * kH + k];
  }
  float* o = out + (size_t)idx * 3;
  o[0] = s0; o[1] = s1; o[2] = s2;
}

// ---------------------------------------------------------------------------
extern "C" void kernel_launch(void* const* d_in, const int* in_sizes, int n_in,
                              void* d_out, int out_size, void* d_ws, size_t ws_size,
                              hipStream_t stream)
{
  const float* x    = (const float*)d_in[0];
  const float* h0   = (const float*)d_in[1];
  const float* c0   = (const float*)d_in[2];
  const float* Wih0 = (const float*)d_in[3];
  const float* Whh0 = (const float*)d_in[4];
  const float* bih0 = (const float*)d_in[5];
  const float* bhh0 = (const float*)d_in[6];
  const float* Wih1 = (const float*)d_in[7];
  const float* Whh1 = (const float*)d_in[8];
  const float* bih1 = (const float*)d_in[9];
  const float* bhh1 = (const float*)d_in[10];
  const float* fcw  = (const float*)d_in[11];
  const float* fcb  = (const float*)d_in[12];
  float* out = (float*)d_out;

  char* ws = (char*)d_ws;
  size_t off = 0;
  auto alloc = [&](size_t bytes) -> char* {
    char* p = ws + off;
    off += (bytes + 255) & ~(size_t)255;
    return p;
  };
  _Float16* wbuf0  = (_Float16*)alloc((size_t)kG * kH * sizeof(_Float16));
  _Float16* wbuf1  = (_Float16*)alloc((size_t)kG * 2 * kH * sizeof(_Float16));
  float*    bsum0  = (float*)alloc(kG * sizeof(float));
  float*    bsum1  = (float*)alloc(kG * sizeof(float));
  float*    wih0c  = (float*)alloc(kG * sizeof(float));
  unsigned* cnts   = (unsigned*)alloc(512);
  _Float16* h0hist = (_Float16*)alloc((size_t)(kT + 1) * kBH * sizeof(_Float16));
  _Float16* h1hist = (_Float16*)alloc((size_t)(kT + 1) * kBH * sizeof(_Float16));

  prep_kernel<<<256, 256, 0, stream>>>(Whh0, Wih1, Whh1, Wih0, bih0, bhh0,
                                       bih1, bhh1, h0, wbuf0, wbuf1,
                                       bsum0, bsum1, wih0c, h0hist, h1hist, cnts);

  const size_t smemFixed = (size_t)(kNCOL * kB + kUPW * kB) * sizeof(float);
  const size_t smem0 = smemFixed + (size_t)kNCOL * kH * sizeof(_Float16);      // ~148 KB
  const size_t smem1 = smemFixed + (size_t)kNCOL * 2 * kH * sizeof(_Float16);  // ~276 KB

  // Layer 0: K = 512 (recurrent only; x-term folded into elementwise).
  lstm_scan_kernel<<<kNWG, 256, smem0, stream>>>(
      wbuf0, bsum0, wih0c, x, nullptr, h0hist, c0, cnts + 0, kH, 0);
  // Layer 1: K = 1024 (fused [W_ih1 | W_hh1], A = [h0_t | h1_{t-1}]).
  lstm_scan_kernel<<<kNWG, 256, smem1, stream>>>(
      wbuf1, bsum1, nullptr, nullptr, h0hist, h1hist, c0 + kBH, cnts + 64,
      2 * kH, 16);

  fc_kernel<<<(kB * kT + 255) / 256, 256, 0, stream>>>(h1hist, fcw, fcb, out);
}
